// MultiHeadAttention_4544075399560
// MI455X (gfx1250) — matrix operand order repaired and checked by reading, not compiled, not run on hardware
//
#include <hip/hip_runtime.h>
#include <hip/hip_bf16.h>

typedef __attribute__((ext_vector_type(16))) _Float16 v16h;
typedef __attribute__((ext_vector_type(8)))  float    v8f;

#define NN    4096
#define EMB   512
#define HEADS 8
#define DHEAD 64

static __device__ __forceinline__ v8f wmma_f16(v16h a, v16h b, v8f c) {
    return __builtin_amdgcn_wmma_f32_16x16x32_f16(false, a, false, b, (short)0, c, false, false);
}

// Async global->LDS copy (gfx1250, tracked by ASYNCcnt). ldsOff is the AS3
// byte offset (low 32 bits of a generic pointer into __shared__ memory).
static __device__ __forceinline__ void async_load_b128(unsigned ldsOff, const void* gptr) {
    asm volatile("global_load_async_to_lds_b128 %0, %1, off"
                 :: "v"(ldsOff), "v"((unsigned long long)(uintptr_t)gptr)
                 : "memory");
}

// ---------------------------------------------------------------------------
// Projection GEMM: out[4096 x 512] (f16) = X[4096x512 f32] @ W[512x512 f32]+b
// One wave computes a 16x16 tile, K-loop in chunks of 32.
// TRANSPOSED==1 writes Vt layout out[col*4096+row] (col = h*64+f -> Vt[h][f][node]).
// ---------------------------------------------------------------------------
template <int TRANSPOSED>
__global__ void proj16(const float* __restrict__ X, const float* __restrict__ W,
                       const float* __restrict__ bias, _Float16* __restrict__ out) {
    const int lane = threadIdx.x & 31;
    const int n    = lane & 15;
    const int hi   = lane >> 4;
    const int row0 = blockIdx.x * 16;
    const int col0 = blockIdx.y * 16;

    v8f acc = {};
    for (int k0 = 0; k0 < EMB; k0 += 32) {
        v16h a, b;
        // A-frag: lane(n,hi): element e -> K = 16*hi + e
        const float* xr = X + (size_t)(row0 + n) * EMB + k0 + 16 * hi;
#pragma unroll
        for (int e = 0; e < 8; ++e) {
            a[e]     = (_Float16)xr[e];
            a[e + 8] = (_Float16)xr[8 + e];
        }
        // B-frag: lane(n,hi): element e -> K = 16*hi + e, N = n
        const float* wc = W + (size_t)(k0 + 16 * hi) * EMB + col0 + n;
#pragma unroll
        for (int e = 0; e < 16; ++e) b[e] = (_Float16)wc[(size_t)e * EMB];
        acc = wmma_f16(a, b, acc);
    }
    const float bb = bias[col0 + n];
#pragma unroll
    for (int g = 0; g < 8; ++g) {
        const int rr = row0 + g + 8 * hi;
        const int cc = col0 + n;
        const float v = acc[g] + bb;
        if constexpr (!TRANSPOSED) out[(size_t)rr * EMB + cc] = (_Float16)v;
        else                       out[(size_t)cc * NN  + rr] = (_Float16)v;
    }
}

// ---------------------------------------------------------------------------
// Fused flash-style attention per head.
// Block = 128 threads (4 waves); each wave owns 16 query rows of one head.
// adj tiles (the 512MB stream) are double-buffered into LDS with async
// global->LDS b128 loads, pipelined one iteration ahead.
// ---------------------------------------------------------------------------
__global__ void attn_fused(const _Float16* __restrict__ Q,
                           const _Float16* __restrict__ K,
                           const _Float16* __restrict__ Vt,
                           const float* __restrict__ adj,
                           const float* __restrict__ alpha,
                           const float* __restrict__ beta,
                           _Float16* __restrict__ Z) {
    __shared__ float    s_S[4][16 * 32];
    __shared__ _Float16 s_P[4][16 * 32];
    __shared__ float    s_scale[4][16];
    __shared__ float    s_O[4][16 * DHEAD];
    __shared__ float    s_adj[4][2][16 * 32];   // double-buffered adj tile

    const int wave = threadIdx.x >> 5;
    const int lane = threadIdx.x & 31;
    const int n    = lane & 15;
    const int hi   = lane >> 4;
    const int h    = blockIdx.y;
    const int row0 = (blockIdx.x * 4 + wave) * 16;

    const float* adjh = adj + ((size_t)h * NN + row0) * (size_t)NN;

    // Issue one adj tile (16 rows x 32 f32 = 2KB): 4 x b128 per lane.
    auto issue_adj = [&](int c0, int buf) {
        const unsigned ldsBase = (unsigned)(uintptr_t)&s_adj[wave][buf][0];
#pragma unroll
        for (int it = 0; it < 4; ++it) {
            const int chunk = lane + it * 32;          // 0..127
            const int row   = chunk >> 3;              // 16 rows
            const int cch   = chunk & 7;               // 8 x 16B per row
            const void* g = (const char*)(adjh + (size_t)row * NN + c0) + cch * 16;
            async_load_b128(ldsBase + chunk * 16, g);
        }
    };

    // Load Q A-fragments once
    v16h qa0, qa1;
    {
        const _Float16* qr = Q + (size_t)(row0 + n) * EMB + h * DHEAD;
#pragma unroll
        for (int e = 0; e < 8; ++e) {
            qa0[e]     = qr[16 * hi + e];
            qa0[e + 8] = qr[8 + 16 * hi + e];
            qa1[e]     = qr[32 + 16 * hi + e];
            qa1[e + 8] = qr[40 + 16 * hi + e];
        }
    }

    v8f o0 = {}, o1 = {}, o2 = {}, o3 = {};
    float m = -__builtin_inff();
    float l = 0.f;
    float dsum = 0.f;

    issue_adj(0, 0);

    for (int c0 = 0; c0 < NN; c0 += 32) {
        const int cur = (c0 >> 5) & 1;

        // ---- S = Q K^T : two 16x16 D-frags (4 WMMAs) ----
        v8f s0, s1;
#pragma unroll
        for (int sb = 0; sb < 2; ++sb) {
            v16h kb0, kb1;
            const _Float16* kr =
                K + (size_t)(c0 + sb * 16 + n) * EMB + h * DHEAD + 16 * hi;
#pragma unroll
            for (int e = 0; e < 16; ++e) { kb0[e] = kr[e]; kb1[e] = kr[32 + e]; }
            v8f acc = {};
            acc = wmma_f16(qa0, kb0, acc);
            acc = wmma_f16(qa1, kb1, acc);
            if (sb == 0) s0 = acc; else s1 = acc;
        }

        // ---- pipeline: issue next adj tile, then wait for current ----
        if (c0 + 32 < NN) {
            issue_adj(c0 + 32, cur ^ 1);
            asm volatile("s_wait_asynccnt 4" ::: "memory");
        } else {
            asm volatile("s_wait_asynccnt 0" ::: "memory");
        }

        // ---- scale + add adj (from LDS), stage S to LDS ----
#pragma unroll
        for (int g = 0; g < 8; ++g) {
            const int rr = g + 8 * hi;
            s_S[wave][rr * 32 + n]      = s0[g] * 0.125f + s_adj[wave][cur][rr * 32 + n];
            s_S[wave][rr * 32 + 16 + n] = s1[g] * 0.125f + s_adj[wave][cur][rr * 32 + 16 + n];
        }
        asm volatile("s_wait_dscnt 0" ::: "memory");

        // ---- online softmax: lanes 0..15 each own one row ----
        if (lane < 16) {
            const int rr = lane;
            float rowv[32];
            float mx = m;
#pragma unroll
            for (int j = 0; j < 32; ++j) {
                rowv[j] = s_S[wave][rr * 32 + j];
                mx = fmaxf(mx, rowv[j]);
            }
            const float fac = __expf(m - mx);
            float psum = 0.f;
            const int diagj = (row0 + rr) - c0;
            float pdiag = 0.f;
#pragma unroll
            for (int j = 0; j < 32; ++j) {
                const float p = __expf(rowv[j] - mx);
                psum += p;
                s_P[wave][rr * 32 + j] = (_Float16)p;
                if (j == diagj) pdiag = p;
            }
            l = l * fac + psum;
            dsum = (diagj >= 0 && diagj < 32) ? pdiag : dsum * fac;
            m = mx;
            s_scale[wave][rr] = fac;
        }
        asm volatile("s_wait_dscnt 0" ::: "memory");

        // ---- rescale O, then O += P @ V (4 WMMAs) ----
#pragma unroll
        for (int g = 0; g < 8; ++g) {
            const float f = s_scale[wave][g + 8 * hi];
            o0[g] *= f; o1[g] *= f; o2[g] *= f; o3[g] *= f;
        }
        v16h pa;
#pragma unroll
        for (int e = 0; e < 8; ++e) {
            pa[e]     = s_P[wave][n * 32 + 16 * hi + e];
            pa[e + 8] = s_P[wave][n * 32 + 8 + 16 * hi + e];
        }
#pragma unroll
        for (int nb = 0; nb < 4; ++nb) {
            const _Float16* vtb =
                Vt + ((size_t)h * DHEAD + nb * 16 + n) * NN + c0 + 16 * hi;
            v16h vb;
#pragma unroll
            for (int e = 0; e < 16; ++e) vb[e] = vtb[e];
            if      (nb == 0) o0 = wmma_f16(pa, vb, o0);
            else if (nb == 1) o1 = wmma_f16(pa, vb, o1);
            else if (nb == 2) o2 = wmma_f16(pa, vb, o2);
            else              o3 = wmma_f16(pa, vb, o3);
        }
    }

    // ---- finalize: z_i = beta*O_i/l + (alpha - beta*d/l) * v_i ----
#pragma unroll
    for (int g = 0; g < 8; ++g) {
        const int rr = g + 8 * hi;
        s_O[wave][rr * DHEAD + 0  + n] = o0[g];
        s_O[wave][rr * DHEAD + 16 + n] = o1[g];
        s_O[wave][rr * DHEAD + 32 + n] = o2[g];
        s_O[wave][rr * DHEAD + 48 + n] = o3[g];
    }
    asm volatile("s_wait_dscnt 0" ::: "memory");
    if (lane < 16) {
        const int rr = lane;
        const float be = beta[h], al = alpha[h];
        const float invl = 1.f / l;
        const float corr = al - be * (dsum * invl);
        const _Float16* vtr = Vt + (size_t)h * DHEAD * NN + (row0 + rr);
        _Float16* zr = Z + (size_t)(row0 + rr) * EMB + h * DHEAD;
#pragma unroll 8
        for (int f = 0; f < DHEAD; ++f) {
            const float vi = (float)vtr[(size_t)f * NN];
            zr[f] = (_Float16)(be * (s_O[wave][rr * DHEAD + f] * invl) + corr * vi);
        }
    }
}

// ---------------------------------------------------------------------------
// Output GEMM: out[4096x512 f32] = Z[f16] @ w_o[512x512 f32] + b_o
// ---------------------------------------------------------------------------
__global__ void out_gemm(const _Float16* __restrict__ Zh,
                         const float* __restrict__ W,
                         const float* __restrict__ bias,
                         float* __restrict__ out) {
    const int lane = threadIdx.x & 31;
    const int n    = lane & 15;
    const int hi   = lane >> 4;
    const int row0 = blockIdx.x * 16;
    const int col0 = blockIdx.y * 16;

    v8f acc = {};
    for (int k0 = 0; k0 < EMB; k0 += 32) {
        v16h a, b;
        const _Float16* zr = Zh + (size_t)(row0 + n) * EMB + k0 + 16 * hi;
#pragma unroll
        for (int e = 0; e < 8; ++e) { a[e] = zr[e]; a[e + 8] = zr[8 + e]; }
        const float* wc = W + (size_t)(k0 + 16 * hi) * EMB + col0 + n;
#pragma unroll
        for (int e = 0; e < 16; ++e) b[e] = (_Float16)wc[(size_t)e * EMB];
        acc = wmma_f16(a, b, acc);
    }
    const float bb = bias[col0 + n];
#pragma unroll
    for (int g = 0; g < 8; ++g)
        out[(size_t)(row0 + g + 8 * hi) * EMB + col0 + n] = acc[g] + bb;
}

// ---------------------------------------------------------------------------
extern "C" void kernel_launch(void* const* d_in, const int* in_sizes, int n_in,
                              void* d_out, int out_size, void* d_ws, size_t ws_size,
                              hipStream_t stream) {
    const float* x     = (const float*)d_in[0];
    const float* adj   = (const float*)d_in[1];
    const float* w_q   = (const float*)d_in[2];
    const float* b_q   = (const float*)d_in[3];
    const float* w_k   = (const float*)d_in[4];
    const float* b_k   = (const float*)d_in[5];
    const float* w_v   = (const float*)d_in[6];
    const float* b_v   = (const float*)d_in[7];
    const float* w_o   = (const float*)d_in[8];
    const float* b_o   = (const float*)d_in[9];
    const float* alpha = (const float*)d_in[10];
    const float* beta  = (const float*)d_in[11];
    float* out = (float*)d_out;

    char* ws = (char*)d_ws;
    const size_t bytesPerMat = (size_t)NN * EMB * sizeof(_Float16); // 4 MB
    _Float16* Qb  = (_Float16*)(ws + 0 * bytesPerMat);
    _Float16* Kb  = (_Float16*)(ws + 1 * bytesPerMat);
    _Float16* Vtb = (_Float16*)(ws + 2 * bytesPerMat);
    _Float16* Zb  = (_Float16*)(ws + 3 * bytesPerMat);

    dim3 gGemm(NN / 16, EMB / 16);
    proj16<0><<<gGemm, 32, 0, stream>>>(x, w_q, b_q, Qb);
    proj16<0><<<gGemm, 32, 0, stream>>>(x, w_k, b_k, Kb);
    proj16<1><<<gGemm, 32, 0, stream>>>(x, w_v, b_v, Vtb);

    dim3 gAttn(NN / 64, HEADS);
    attn_fused<<<gAttn, 128, 0, stream>>>(Qb, Kb, Vtb, adj, alpha, beta, Zb);

    out_gemm<<<gGemm, 32, 0, stream>>>(Zb, w_o, b_o, out);
}
